// LSTMRegressor_18425409700120
// MI455X (gfx1250) — matrix operand order repaired and checked by reading, not compiled, not run on hardware
//
#include <hip/hip_runtime.h>

// ---------------------------------------------------------------------------
// LSTM regressor for MI455X (gfx1250, wave32, WMMA).
//
// Decomposition: B=256 independent recurrences -> 16 workgroups x 16 batches,
// each workgroup runs all T=512 steps with only intra-WG barriers (no global
// sync). Per step, per WG: GEMM  gates[16 x 2048] = [x_t|h][16 x 640] @ W^T,
// via v_wmma_f32_16x16x32_f16 (f16 in, f32 accumulate). Weights (2.6 MB f16)
// are L2-resident (192 MB L2); h is kept in LDS (f16); c stays in VGPRs in
// WMMA C-layout for the entire kernel.
//
// NOTE on the asm laundering: without it, LICM proves the 160 weight
// B-fragments loop-invariant across t, hoists 1280 VGPRs of weights and
// spills them to scratch (round-1 disasm). Laundering the *base pointer*
// through an integer destroys address-space provenance and yields
// flat_load_b128 (round-2 disasm), whose dual LOADcnt+DScnt accounting
// cross-serializes weights with LDS traffic. So we launder a 32-bit zero
// OFFSET instead: provenance stays on the kernarg (-> global_load, SADDR
// form) but every address still depends on an opaque per-iteration value.
// ---------------------------------------------------------------------------

#define B_TOT    256
#define T_STEPS  512
#define F_IN     128
#define H_DIM    512
#define N_GATES  2048         // 4*H
#define K_TOT    640          // F + H
#define ROWPAD   648          // LDS row stride in halves (pad: 64-bank friendly)

typedef __attribute__((ext_vector_type(16))) _Float16     v16h;
typedef __attribute__((ext_vector_type(4)))  _Float16     v4h;
typedef __attribute__((ext_vector_type(8)))  float        v8f;
typedef __attribute__((ext_vector_type(4)))  unsigned int u32x4;

union Frag16 { v16h v; u32x4 q[2]; };

__device__ __forceinline__ float sigmoid_f(float x) {
    return 1.0f / (1.0f + __expf(-x));
}

// ---------------------------------------------------------------------------
// Prep: Wcat[n][k] (f16, row-major, k = [x cols 0..127 | h cols 128..639]),
//       bsum[n] = b_ih[n] + b_hh[n]  (f32).
// ---------------------------------------------------------------------------
__global__ void prep_kernel(const float* __restrict__ W_ih,
                            const float* __restrict__ W_hh,
                            const float* __restrict__ b_ih,
                            const float* __restrict__ b_hh,
                            _Float16*    __restrict__ Wcat,
                            float*       __restrict__ bsum)
{
    int i = blockIdx.x * blockDim.x + threadIdx.x;    // exactly 2048*640 threads
    int row = i / K_TOT;
    int col = i - row * K_TOT;
    float v = (col < F_IN) ? W_ih[row * F_IN + col]
                           : W_hh[row * H_DIM + (col - F_IN)];
    Wcat[i] = (_Float16)v;
    if (i < N_GATES) bsum[i] = b_ih[i] + b_hh[i];
}

// ---------------------------------------------------------------------------
// Persistent LSTM: grid = 16 WGs (one per 16-batch group), block = 512 (16
// waves). Wave w owns h-column tiles {w, w+16} -> 8 gate tiles {ht + 32*g}.
// LDS holds xh[16][648] f16: cols 0..127 = x_t, cols 128..639 = h.
// ---------------------------------------------------------------------------
__global__ __launch_bounds__(512, 1) void lstm_kernel(
    const float*    __restrict__ x,       // [B, T, F] f32
    const _Float16* __restrict__ Wcat,    // [2048, 640] f16
    const float*    __restrict__ bsum,    // [2048] f32
    float*          __restrict__ hout)    // [B, H] f32 (final h)
{
    __shared__ _Float16 xh[16 * ROWPAD];  // 20.7 KB

    const int tid   = threadIdx.x;
    const int wave  = tid >> 5;           // 0..15
    const int lane  = tid & 31;
    const int bbase = blockIdx.x * 16;

    const int lrow  = lane & 15;          // matrix row/col within half-wave
    const int lsel  = lane >> 4;          // 0 | 1 (selects lane half)

    // zero the h region of LDS (cols 128..639)
    for (int i = tid; i < 16 * H_DIM; i += 512) {
        int r = i >> 9;
        int c = i & (H_DIM - 1);
        xh[r * ROWPAD + F_IN + c] = (_Float16)0.0f;
    }

    const int ht[2] = { wave, wave + 16 };

    // per-tile element offsets into Wcat: tile j = u*4+g -> n-tile ht[u]+32*g.
    // B-frag (32x16, 16-bit): lane holds column n = n-tile*16 + lrow, i.e.
    // weight row n; lanes 0-15 take K {kb..kb+7, kb+16..23}, lanes 16-31 +8.
    unsigned int rowElem[8];
#pragma unroll
    for (int u = 0; u < 2; ++u)
#pragma unroll
        for (int g = 0; g < 4; ++g) {
            int nt = ht[u] + 32 * g;
            rowElem[u * 4 + g] =
                (unsigned int)(nt * 16 + lrow) * K_TOT + lsel * 8;
        }

    // biases: per-lane scalar per gate tile (column = ht*16 + lrow)
    float bias[8];
#pragma unroll
    for (int u = 0; u < 2; ++u)
#pragma unroll
        for (int g = 0; g < 4; ++g)
            bias[u * 4 + g] = bsum[g * H_DIM + ht[u] * 16 + lrow];

    // persistent cell state + last h, in WMMA C-layout (M = r + 8*lsel, N = lrow)
    v8f cst[2], hfin[2];
#pragma unroll
    for (int u = 0; u < 2; ++u)
#pragma unroll
        for (int r = 0; r < 8; ++r) { cst[u][r] = 0.0f; hfin[u][r] = 0.0f; }

    // x staging map: 512 threads cover 16 rows x 128 cols (float4 each)
    const int xr = tid >> 5;              // batch row 0..15
    const int xc = (tid & 31) * 4;        // col 0..124
    const float* xrow = x + (size_t)(bbase + xr) * T_STEPS * F_IN + xc;

    // A-frag (16x32, 16-bit): lane holds row lrow, K {kb..kb+7} and
    // {kb+16..kb+23} (lanes 16-31 shifted by +8 halves).
    const char* aptr = (const char*)&xh[lrow * ROWPAD] + lsel * 16;

    for (int t = 0; t < T_STEPS; ++t) {
        // [A] stage x_t (f32 -> f16) into LDS cols 0..127
        float4 xv = *(const float4*)(xrow + (size_t)t * F_IN);
        v4h xp = { (_Float16)xv.x, (_Float16)xv.y, (_Float16)xv.z, (_Float16)xv.w };
        *(v4h*)&xh[xr * ROWPAD + xc] = xp;
        __syncthreads();                  // [B] x + previous-step h visible

        // opaque zero offset: keeps kernarg provenance (global_load) but
        // defeats LICM hoisting of the weight fragments across t.
        unsigned int zoff = 0;
        asm volatile("" : "+v"(zoff));
        const _Float16* wbL = Wcat + zoff;
        const _Float16* wp[8];
#pragma unroll
        for (int j = 0; j < 8; ++j)
            wp[j] = wbL + rowElem[j];

        // [C] GEMM: 8 gate tiles, K = 640 in 20 chunks of 32
        v8f acc[8];
#pragma unroll
        for (int j = 0; j < 8; ++j)
#pragma unroll
            for (int r = 0; r < 8; ++r) acc[j][r] = 0.0f;

#pragma unroll
        for (int kc = 0; kc < K_TOT / 32; ++kc) {
            const int kb = kc * 32;       // K-chunk base (halves)
            Frag16 a;
            a.q[0] = *(const u32x4*)((const char*)aptr + kb * 2);      // ds_load_b128
            a.q[1] = *(const u32x4*)((const char*)aptr + kb * 2 + 32);
            Frag16 bfr[8];
#pragma unroll
            for (int j = 0; j < 8; ++j) {                // 16x global_load_b128
                bfr[j].q[0] = *(const u32x4*)(wp[j] + kb);
                bfr[j].q[1] = *(const u32x4*)(wp[j] + kb + 16);
            }
#pragma unroll
            for (int j = 0; j < 8; ++j)                  // 8 independent chains
                acc[j] = __builtin_amdgcn_wmma_f32_16x16x32_f16(
                    false, a.v, false, bfr[j].v, (short)0, acc[j], false, false);
        }
        __syncthreads();                  // [D] all LDS reads of x|h done

        // [E] gate nonlinearity + cell/h update; write new h (f16) to LDS
#pragma unroll
        for (int u = 0; u < 2; ++u) {
            const int colbase = F_IN + ht[u] * 16 + lrow;
#pragma unroll
            for (int r = 0; r < 8; ++r) {
                float iv = sigmoid_f(acc[u * 4 + 0][r] + bias[u * 4 + 0]);
                float fv = sigmoid_f(acc[u * 4 + 1][r] + bias[u * 4 + 1]);
                float gv = tanhf    (acc[u * 4 + 2][r] + bias[u * 4 + 2]);
                float ov = sigmoid_f(acc[u * 4 + 3][r] + bias[u * 4 + 3]);
                float cv = fv * cst[u][r] + iv * gv;
                cst[u][r] = cv;
                float hv = ov * tanhf(cv);
                hfin[u][r] = hv;
                xh[(r + lsel * 8) * ROWPAD + colbase] = (_Float16)hv;
            }
        }
        // next iteration's barrier [B] publishes these h stores before reads;
        // [A] writes (cols 0..127) are disjoint from h (cols 128..639).
    }

    // write final h (f32, full precision) for the MLP head
#pragma unroll
    for (int u = 0; u < 2; ++u) {
        const int col = ht[u] * 16 + lrow;
#pragma unroll
        for (int r = 0; r < 8; ++r) {
            int brow = bbase + r + lsel * 8;
            hout[(size_t)brow * H_DIM + col] = hfin[u][r];
        }
    }
}

// ---------------------------------------------------------------------------
// Head: out = ((relu(h@W1^T+b1) -> relu(@W2^T+b2)) @ W3^T + b3), [256] f32.
// One thread per batch row; W1 staged in LDS. ~1.6 MFLOP total.
// ---------------------------------------------------------------------------
__global__ void head_kernel(const float* __restrict__ h,
                            const float* __restrict__ W1, const float* __restrict__ b1,
                            const float* __restrict__ W2, const float* __restrict__ b2,
                            const float* __restrict__ W3, const float* __restrict__ b3,
                            float* __restrict__ out)
{
    __shared__ float w1s[12 * H_DIM];
    for (int i = threadIdx.x; i < 12 * H_DIM; i += blockDim.x) w1s[i] = W1[i];
    __syncthreads();

    int b = blockIdx.x * blockDim.x + threadIdx.x;
    if (b >= B_TOT) return;
    const float* hb = h + (size_t)b * H_DIM;

    float a1[12];
#pragma unroll
    for (int j = 0; j < 12; ++j) {
        float s = b1[j];
        for (int k = 0; k < H_DIM; ++k) s += hb[k] * w1s[j * H_DIM + k];
        a1[j] = fmaxf(s, 0.0f);
    }
    float a2[12];
#pragma unroll
    for (int j = 0; j < 12; ++j) {
        float s = b2[j];
#pragma unroll
        for (int k = 0; k < 12; ++k) s += a1[k] * W2[j * 12 + k];
        a2[j] = fmaxf(s, 0.0f);
    }
    float s = b3[0];
#pragma unroll
    for (int k = 0; k < 12; ++k) s += a2[k] * W3[k];
    out[b] = s;
}

// ---------------------------------------------------------------------------
extern "C" void kernel_launch(void* const* d_in, const int* in_sizes, int n_in,
                              void* d_out, int out_size, void* d_ws, size_t ws_size,
                              hipStream_t stream)
{
    const float* x    = (const float*)d_in[0];
    const float* W_ih = (const float*)d_in[1];
    const float* W_hh = (const float*)d_in[2];
    const float* b_ih = (const float*)d_in[3];
    const float* b_hh = (const float*)d_in[4];
    const float* W1   = (const float*)d_in[5];
    const float* b1   = (const float*)d_in[6];
    const float* W2   = (const float*)d_in[7];
    const float* b2   = (const float*)d_in[8];
    const float* W3   = (const float*)d_in[9];
    const float* b3   = (const float*)d_in[10];
    float* out = (float*)d_out;

    // workspace layout (all 16B aligned): Wcat f16 | bsum f32 | hbuf f32
    char* ws = (char*)d_ws;
    _Float16* Wcat = (_Float16*)ws;                                   // 2,621,440 B
    float*    bsum = (float*)(ws + (size_t)N_GATES * K_TOT * 2);      //     8,192 B
    float*    hbuf = (float*)(ws + (size_t)N_GATES * K_TOT * 2
                                 + (size_t)N_GATES * 4);              //   524,288 B

    prep_kernel<<<(N_GATES * K_TOT) / 256, 256, 0, stream>>>(
        W_ih, W_hh, b_ih, b_hh, Wcat, bsum);
    lstm_kernel<<<B_TOT / 16, 512, 0, stream>>>(x, Wcat, bsum, hbuf);
    head_kernel<<<1, 256, 0, stream>>>(hbuf, W1, b1, W2, b2, W3, b3, out);
}
